// VariationalLinearEncoder_6760278524377
// MI455X (gfx1250) — matrix operand order repaired and checked by reading, not compiled, not run on hardware
//
#include <hip/hip_runtime.h>

#define N_NODES 50000
#define N_EDGES 800000
#define IN_C    512
#define OUT_C   128
#define HC      256   // combined output channels (mu | logstd)

typedef __attribute__((ext_vector_type(16))) __bf16 v16bf;
typedef __attribute__((ext_vector_type(8)))  float  v8f;

// ---------- helpers ----------
static __device__ __forceinline__ unsigned short f2bf_bits(float f) {
    unsigned u = __builtin_bit_cast(unsigned, f);
    unsigned r = u + 0x7FFFu + ((u >> 16) & 1u);   // round to nearest even
    return (unsigned short)(r >> 16);
}
static __device__ __forceinline__ __bf16 f2bf(float f) {
    unsigned short b = f2bf_bits(f);
    return __builtin_bit_cast(__bf16, b);
}

// ---------- 1) build Wt[256][512] bf16 = transpose of [W_mu | W_logstd] ----------
__global__ void prep_w(const float* __restrict__ Wmu, const float* __restrict__ Wls,
                       unsigned short* __restrict__ Wt) {
    int idx = blockIdx.x * blockDim.x + threadIdx.x;     // 256*512 total
    if (idx >= HC * IN_C) return;
    int n = idx >> 9;         // / 512
    int k = idx & (IN_C - 1); // % 512
    float v = (n < OUT_C) ? Wmu[k * OUT_C + n] : Wls[k * OUT_C + (n - OUT_C)];
    Wt[(size_t)n * IN_C + k] = f2bf_bits(v);
}

// ---------- 2) degree with self-loops, then dinv ----------
__global__ void deg_init(float* __restrict__ deg) {
    int i = blockIdx.x * blockDim.x + threadIdx.x;
    if (i < N_NODES) deg[i] = 1.0f;                      // self-loop contributes 1
}
__global__ void deg_scatter(const long long* __restrict__ dst, float* __restrict__ deg) {
    int e = blockIdx.x * blockDim.x + threadIdx.x;
    if (e < N_EDGES) atomicAdd(&deg[(int)dst[e]], 1.0f);
}
__global__ void dinv_kernel(float* __restrict__ deg) {
    int i = blockIdx.x * blockDim.x + threadIdx.x;
    if (i < N_NODES) deg[i] = rsqrtf(deg[i]);            // deg >= 1 always
}

// ---------- 3) WMMA GEMM: h[50000][256] = x[50000][512] @ Wt^T ----------
// Block: 256 threads = 8 waves. Waves 0-3 -> rows [blk*32 .. +15], waves 4-7 -> +16..31.
// Wave (w&3) covers columns (w&3)*64 .. +63 as four 16x16 tiles.
__global__ void gemm_wmma(const float* __restrict__ x,
                          const unsigned short* __restrict__ Wt,
                          float* __restrict__ h) {
    const int lane   = threadIdx.x & 31;
    const int wave   = threadIdx.x >> 5;
    const int rowTile = blockIdx.x * 32 + (wave >> 2) * 16;
    const int colBase = (wave & 3) * 64;
    const int half = lane >> 4;     // 0: lanes 0-15, 1: lanes 16-31
    const int l16  = lane & 15;

    int arow = rowTile + l16;
    if (arow >= N_NODES) arow = N_NODES - 1;             // clamp; stores are masked
    const float* xr = x + (size_t)arow * IN_C;

    v8f acc0 = {}, acc1 = {}, acc2 = {}, acc3 = {};

    for (int k0 = 0; k0 < IN_C; k0 += 32) {
        // A-matrix (16x32 bf16): lanes 0-15 hold K=k0+0..15,
        // lanes 16-31 hold K=k0+16..31 (same contiguous K order as B).
        const int klo = k0 + half * 16;
        float4 f0 = *(const float4*)(xr + klo);
        float4 f1 = *(const float4*)(xr + klo + 4);
        float4 f2 = *(const float4*)(xr + klo + 8);
        float4 f3 = *(const float4*)(xr + klo + 12);
        v16bf a;
        a[0]  = f2bf(f0.x); a[1]  = f2bf(f0.y); a[2]  = f2bf(f0.z); a[3]  = f2bf(f0.w);
        a[4]  = f2bf(f1.x); a[5]  = f2bf(f1.y); a[6]  = f2bf(f1.z); a[7]  = f2bf(f1.w);
        a[8]  = f2bf(f2.x); a[9]  = f2bf(f2.y); a[10] = f2bf(f2.z); a[11] = f2bf(f2.w);
        a[12] = f2bf(f3.x); a[13] = f2bf(f3.y); a[14] = f2bf(f3.z); a[15] = f2bf(f3.w);

        // B-matrix (32x16 bf16): lane holds 16 consecutive K for its column;
        // lanes 0-15 -> K=k0..k0+15, lanes 16-31 -> K=k0+16..k0+31.
        const int kb = k0 + half * 16;
        const v16bf b0 = *(const v16bf*)(Wt + (size_t)(colBase +  0 + l16) * IN_C + kb);
        const v16bf b1 = *(const v16bf*)(Wt + (size_t)(colBase + 16 + l16) * IN_C + kb);
        const v16bf b2 = *(const v16bf*)(Wt + (size_t)(colBase + 32 + l16) * IN_C + kb);
        const v16bf b3 = *(const v16bf*)(Wt + (size_t)(colBase + 48 + l16) * IN_C + kb);

        acc0 = __builtin_amdgcn_wmma_f32_16x16x32_bf16(false, a, false, b0, (short)0, acc0, false, false);
        acc1 = __builtin_amdgcn_wmma_f32_16x16x32_bf16(false, a, false, b1, (short)0, acc1, false, false);
        acc2 = __builtin_amdgcn_wmma_f32_16x16x32_bf16(false, a, false, b2, (short)0, acc2, false, false);
        acc3 = __builtin_amdgcn_wmma_f32_16x16x32_bf16(false, a, false, b3, (short)0, acc3, false, false);
    }

    // D layout: lanes 0-15 -> M = vgpr r, lanes 16-31 -> M = 8 + r; N = lane&15.
    const int mBase = rowTile + half * 8;
#pragma unroll
    for (int r = 0; r < 8; ++r) {
        int m = mBase + r;
        if (m < N_NODES) {
            float* hr = h + (size_t)m * HC;
            hr[colBase +  0 + l16] = acc0[r];
            hr[colBase + 16 + l16] = acc1[r];
            hr[colBase + 32 + l16] = acc2[r];
            hr[colBase + 48 + l16] = acc3[r];
        }
    }
}

// ---------- 4) out = bias + dinv[n]^2 * h[n]  (self-loop term; initializes d_out) ----------
__global__ void out_init(const float* __restrict__ h, const float* __restrict__ dinv,
                         const float* __restrict__ bmu, const float* __restrict__ bls,
                         float* __restrict__ out) {
    long long idx = (long long)blockIdx.x * blockDim.x + threadIdx.x;
    if (idx >= (long long)N_NODES * HC) return;
    int n = (int)(idx >> 8);       // / 256
    int c = (int)(idx & 255);
    float w = dinv[n] * dinv[n];
    float v = h[(size_t)n * HC + c] * w;
    if (c < OUT_C)
        out[(size_t)n * OUT_C + c] = bmu[c] + v;
    else
        out[(size_t)N_NODES * OUT_C + (size_t)n * OUT_C + (c - OUT_C)] = bls[c - OUT_C] + v;
}

// ---------- 5) edge scatter: 64 threads/edge, 4 channels each ----------
__global__ void edge_scatter(const long long* __restrict__ src, const long long* __restrict__ dst,
                             const float* __restrict__ h, const float* __restrict__ dinv,
                             float* __restrict__ out) {
    long long t = (long long)blockIdx.x * blockDim.x + threadIdx.x;
    if (t >= (long long)N_EDGES * 64) return;
    int e  = (int)(t >> 6);
    int c4 = (int)(t & 63) << 2;                 // channel base in [0,252]
    int s  = (int)src[e];
    int d  = (int)dst[e];
    float norm = dinv[s] * dinv[d];
    float4 hv = *(const float4*)(h + (size_t)s * HC + c4);
    float* base = (c4 < OUT_C)
        ? (out + (size_t)d * OUT_C + c4)
        : (out + (size_t)N_NODES * OUT_C + (size_t)d * OUT_C + (c4 - OUT_C));
    atomicAdd(base + 0, hv.x * norm);
    atomicAdd(base + 1, hv.y * norm);
    atomicAdd(base + 2, hv.z * norm);
    atomicAdd(base + 3, hv.w * norm);
}

extern "C" void kernel_launch(void* const* d_in, const int* in_sizes, int n_in,
                              void* d_out, int out_size, void* d_ws, size_t ws_size,
                              hipStream_t stream) {
    const float*     x   = (const float*)d_in[0];
    const long long* ei  = (const long long*)d_in[1];   // int64 [2,800000]
    const float*     Wmu = (const float*)d_in[2];
    const float*     bmu = (const float*)d_in[3];
    const float*     Wls = (const float*)d_in[4];
    const float*     bls = (const float*)d_in[5];
    float* out = (float*)d_out;

    const long long* src = ei;
    const long long* dst = ei + N_EDGES;

    // workspace layout
    char* ws = (char*)d_ws;
    float*          h   = (float*)ws;                                   // 51,200,000 B
    float*          deg = (float*)(ws + 51200000);                      //    204,800 B (padded)
    unsigned short* Wt  = (unsigned short*)(ws + 51200000 + 204800);    //    262,144 B

    // 1) weight transpose + bf16 convert
    prep_w<<<(HC * IN_C + 255) / 256, 256, 0, stream>>>(Wmu, Wls, Wt);

    // 2) degrees -> dinv
    deg_init<<<(N_NODES + 255) / 256, 256, 0, stream>>>(deg);
    deg_scatter<<<(N_EDGES + 255) / 256, 256, 0, stream>>>(dst, deg);
    dinv_kernel<<<(N_NODES + 255) / 256, 256, 0, stream>>>(deg);

    // 3) fused GEMM h = x @ [W_mu | W_logstd] via bf16 WMMA
    gemm_wmma<<<(N_NODES + 31) / 32, 256, 0, stream>>>(x, Wt, h);

    // 4) initialize outputs with bias + self-loop term
    {
        long long total = (long long)N_NODES * HC;
        out_init<<<(unsigned)((total + 255) / 256), 256, 0, stream>>>(h, deg, bmu, bls, out);
    }

    // 5) scatter messages over edges
    {
        long long total = (long long)N_EDGES * 64;
        edge_scatter<<<(unsigned)((total + 255) / 256), 256, 0, stream>>>(src, dst, h, deg, out);
    }
}
